// TomoGNN_8521215115529
// MI455X (gfx1250) — compile-verified, matrix operand order checked by reading, not run on hardware
//
#include <hip/hip_runtime.h>
#include <hip/hip_bf16.h>

typedef __attribute__((ext_vector_type(16))) _Float16 v16h;
typedef __attribute__((ext_vector_type(8)))  float    v8f;

#define NSN 16384           // sino nodes (128*128)
#define HW  16384           // 128*128 spatial
#define BATCH 4

// ---------------------------------------------------------------------------
// Generic implicit-GEMM conv (ks x ks, pad ks/2, H=W=128) via WMMA f16->f32.
// One wave -> 16 spatial pixels (a row strip) x 16 output channels.
// Block = (CoutPad/16) waves sharing an LDS-staged input footprint.
// Weights are pre-packed in WMMA B-fragment order: [kstep][ntile][lane][16],
// so each lane loads its fragment as one 32-byte vector (2x b128).
// mode 0: f16 NCHW out | 1: f32 NCHW out | 2: f32 [b][pix][8] node features
// mode 3: f32 final output [b][1][128][128]
// ---------------------------------------------------------------------------
template<int KS>
__global__ void conv_wmma_k(const _Float16* __restrict__ act,
                            const v16h* __restrict__ wfrag,   // fragment-packed weights
                            const float* __restrict__ bias,
                            const float* __restrict__ prelu,
                            int Cin, int Cout, int CoutPad, int mode,
                            _Float16* __restrict__ outH,
                            float* __restrict__ outF)
{
    constexpr int PAD = KS / 2;
    constexpr int FW  = 16 + KS - 1;
    constexpr int K2  = KS * KS;
    extern __shared__ _Float16 smem[];

    const int x0   = blockIdx.x << 4;
    const int y    = blockIdx.y;
    const int b    = blockIdx.z;
    const int tid  = threadIdx.x;
    const int lane = tid & 31;
    const int wid  = tid >> 5;
    const int NT   = CoutPad >> 4;

    // ---- stage input footprint (x-fastest => coalesced global reads) ----
    // LDS layout: smem[(dy*FW + dx)*Cin + c]
    const int total = KS * FW * Cin;
    for (int i = tid; i < total; i += blockDim.x) {
        int dx  = i % FW;           // compile-time divisor
        int rem = i / FW;
        int dy  = rem % KS;         // compile-time divisor
        int c   = rem / KS;
        int yy  = y + dy - PAD;
        int xx  = x0 + dx - PAD;
        _Float16 v = (_Float16)0.f;
        if (yy >= 0 && yy < 128 && xx >= 0 && xx < 128)
            v = act[(((b * Cin + c) << 14) + (yy << 7)) + xx];
        smem[(dy * FW + dx) * Cin + c] = v;
    }
    __syncthreads();

    const int m     = lane & 15;   // A: M index / D: N index
    const int half  = lane >> 4;   // selects K sub-range per 16-bit layout
    const int Ktot  = Cin * K2;

    v8f acc = {};
    for (int kb = 0; kb < Ktot; kb += 32) {
        const size_t fidx = ((size_t)(kb >> 5) * NT + wid) * 32 + lane;
        if (kb + 32 < Ktot)
            __builtin_prefetch(wfrag + fidx + (size_t)NT * 32, 0, 1);
        // B fragment: single 32B vector load (lowered to 2x global_load_b128)
        v16h bf = wfrag[fidx];
        // A fragment: gather from LDS footprint
        v16h af;
        #pragma unroll
        for (int e = 0; e < 16; ++e) {
            // 16-bit fragment layout: K = e + 8*half (e<8) | e + 8 + 8*half (e>=8)
            int koff = (e < 8) ? e : (e + 8);
            int kg   = kb + koff + (half << 3);
            _Float16 av = (_Float16)0.f;
            if (kg < Ktot) {
                int cin = kg / K2;        // compile-time divisors -> mul/shift
                int r   = kg - cin * K2;
                int kh  = r / KS;
                int kw  = r - kh * KS;
                av = smem[(kh * FW + (m + kw)) * Cin + cin];
            }
            af[e] = av;
        }
        acc = __builtin_amdgcn_wmma_f32_16x16x32_f16(
                  false, af, false, bf, (short)0, acc, false, false);
    }

    // ---- epilogue: bias + PReLU, store per D layout (M = r + 8*half) ----
    const int   ch = (wid << 4) + m;
    const float pa = prelu[0];
    const float bs = (ch < Cout) ? bias[ch] : 0.f;
    #pragma unroll
    for (int r = 0; r < 8; ++r) {
        int mm = r + (half << 3);
        int x  = x0 + mm;
        float v = acc[r] + bs;
        v = (v >= 0.f) ? v : pa * v;
        if (ch < Cout) {
            if (mode == 0)
                outH[(((b * Cout + ch) << 14) + (y << 7)) + x] = (_Float16)v;
            else if (mode == 1)
                outF[(((b * Cout + ch) << 14) + (y << 7)) + x] = v;
            else if (mode == 2)
                outF[(((b << 14) + (y << 7) + x) << 3) + ch] = v;
            else
                outF[(b << 14) + (y << 7) + x] = v;
        }
    }
}

// ---- pack OIHW f32 weights -> WMMA B-fragment order f16 -------------------
// layout: [kstep][ntile][lane(32)][e(16)]; zero-pad kg>=Ktot and n>=Cout
__global__ void pack_w_frag_k(const float* __restrict__ w, _Float16* __restrict__ wp,
                              int Ktot, int Cout, int NT, int totalFrag)
{
    int i = blockIdx.x * blockDim.x + threadIdx.x;
    if (i >= totalFrag) return;
    int e    = i & 15;
    int lane = (i >> 4) & 31;
    int fi   = i >> 9;              // kstep*NT + ntile
    int nt   = fi % NT;
    int ks   = fi / NT;
    int half = lane >> 4;
    int n    = (nt << 4) + (lane & 15);
    int koff = (e < 8) ? e : (e + 8);
    int kg   = (ks << 5) + koff + (half << 3);
    _Float16 v = (_Float16)0.f;
    if (kg < Ktot && n < Cout) v = (_Float16)w[(size_t)n * Ktot + kg];
    wp[i] = v;
}

__global__ void cast_f16_k(const float* __restrict__ in, _Float16* __restrict__ out, int n)
{
    int i = blockIdx.x * blockDim.x + threadIdx.x;
    if (i < n) out[i] = (_Float16)in[i];
}

// ---- BatchNorm2d training-mode stats over (N,H,W), 64 channels ------------
__global__ void bn_reduce_k(const float* __restrict__ x, float* __restrict__ mv)
{
    int c = blockIdx.x;
    float s = 0.f, sq = 0.f;
    for (int i = threadIdx.x; i < BATCH * HW; i += 256) {
        int b  = i >> 14;
        int hw = i & 16383;
        float v = x[((b * 64 + c) << 14) + hw];
        s += v; sq += v * v;
    }
    __shared__ float ss[256], sqq[256];
    ss[threadIdx.x] = s; sqq[threadIdx.x] = sq;
    __syncthreads();
    for (int off = 128; off > 0; off >>= 1) {
        if (threadIdx.x < off) {
            ss[threadIdx.x]  += ss[threadIdx.x + off];
            sqq[threadIdx.x] += sqq[threadIdx.x + off];
        }
        __syncthreads();
    }
    if (threadIdx.x == 0) {
        float m = ss[0]  * (1.f / 65536.f);
        float v = sqq[0] * (1.f / 65536.f) - m * m;
        mv[c] = m; mv[64 + c] = v;
    }
}

__global__ void bn_norm_k(const float* __restrict__ x, const float* __restrict__ mv,
                          const float* __restrict__ g, const float* __restrict__ bb,
                          _Float16* __restrict__ out)
{
    int i = blockIdx.x * blockDim.x + threadIdx.x;
    if (i >= BATCH * 64 * HW) return;
    int c = (i >> 14) & 63;
    float y = (x[i] - mv[c]) * rsqrtf(mv[64 + c] + 1e-5f) * g[c] + bb[c];
    out[i] = (_Float16)y;
}

// ---- GCN: node linear 8->10 (sino nodes only; pixel-node features are 0) --
__global__ void node_lin_k(const float* __restrict__ feat, const float* __restrict__ gw,
                           float* __restrict__ xl)
{
    int idx = blockIdx.x * blockDim.x + threadIdx.x;   // b*NSN + node
    if (idx >= BATCH * NSN) return;
    float f[8];
    #pragma unroll
    for (int k = 0; k < 8; ++k) f[k] = feat[(idx << 3) + k];
    #pragma unroll
    for (int j = 0; j < 10; ++j) {
        float s = 0.f;
        #pragma unroll
        for (int k = 0; k < 8; ++k) s += f[k] * gw[k * 10 + j];
        xl[idx * 10 + j] = s;
    }
}

__global__ void deg_init_k(float* __restrict__ deg)
{
    int i = blockIdx.x * blockDim.x + threadIdx.x;
    if (i < 2 * NSN) deg[i] = 1.0f;                    // self-loop weight
}

__global__ void deg_acc_k(const int* __restrict__ ei, const float* __restrict__ ew,
                          float* __restrict__ deg, int E)
{
    int e = blockIdx.x * blockDim.x + threadIdx.x;
    if (e < E) atomicAdd(&deg[ei[E + e]], ew[e]);      // scatter at targets (col)
}

__global__ void dis_k(const float* __restrict__ deg, float* __restrict__ dis)
{
    int i = blockIdx.x * blockDim.x + threadIdx.x;
    if (i >= 2 * NSN) return;
    float d = deg[i];
    dis[i] = (d > 0.f) ? rsqrtf(fmaxf(d, 1e-30f)) : 0.f;
}

__global__ void zero_k(float* __restrict__ p, int n)
{
    int i = blockIdx.x * blockDim.x + threadIdx.x;
    if (i < n) p[i] = 0.f;
}

// one thread per (edge, batch); only sino->pixel edges carry nonzero messages
__global__ void scatter_k(const int* __restrict__ ei, const float* __restrict__ ew,
                          const float* __restrict__ dis, const float* __restrict__ xl,
                          float* __restrict__ agg, int E)
{
    int t = blockIdx.x * blockDim.x + threadIdx.x;
    if (t >= 4 * E) return;
    int e = t >> 2;
    int b = t & 3;
    int r = ei[e];
    int c = ei[E + e];
    if (r >= NSN || c < NSN) return;                   // source zero / target dropped
    float norm = dis[r] * ew[e] * dis[c];
    int pix = c - NSN;
    const float* xr = xl + ((size_t)(b * NSN + r)) * 10;
    #pragma unroll
    for (int f = 0; f < 10; ++f)
        atomicAdd(&agg[((b * 10 + f) << 14) + pix], xr[f] * norm);
}

__global__ void img_cast_k(const float* __restrict__ agg, const float* __restrict__ gb,
                           _Float16* __restrict__ img)
{
    int i = blockIdx.x * blockDim.x + threadIdx.x;
    if (i >= BATCH * 10 * HW) return;
    int f = (i >> 14) % 10;
    img[i] = (_Float16)(agg[i] + gb[f]);
}

// ---------------------------------------------------------------------------
extern "C" void kernel_launch(void* const* d_in, const int* in_sizes, int n_in,
                              void* d_out, int out_size, void* d_ws, size_t ws_size,
                              hipStream_t stream)
{
    const float* sino   = (const float*)d_in[0];
    const int*   ei     = (const int*)  d_in[1];
    const float* ew     = (const float*)d_in[2];
    const float* af_w1  = (const float*)d_in[3];
    const float* af_w2  = (const float*)d_in[4];
    const float* af_w3  = (const float*)d_in[5];
    const float* af_w4  = (const float*)d_in[6];
    const float* af_w5  = (const float*)d_in[7];
    const float* gcn_w  = (const float*)d_in[8];
    const float* sr_w1  = (const float*)d_in[9];
    const float* sr_w2  = (const float*)d_in[10];
    const float* sr_w3  = (const float*)d_in[11];
    const float* af_b1  = (const float*)d_in[12];
    const float* af_b2  = (const float*)d_in[13];
    const float* af_b3  = (const float*)d_in[14];
    const float* af_b4  = (const float*)d_in[15];
    const float* af_b5  = (const float*)d_in[16];
    const float* sr_b1  = (const float*)d_in[17];
    const float* sr_b2  = (const float*)d_in[18];
    const float* sr_b3  = (const float*)d_in[19];
    const float* gcn_b  = (const float*)d_in[20];
    const float* bn_g   = (const float*)d_in[21];
    const float* bn_b   = (const float*)d_in[22];
    const float* af_p1  = (const float*)d_in[23];
    const float* af_p2  = (const float*)d_in[24];
    const float* af_p3  = (const float*)d_in[25];
    const float* af_p4  = (const float*)d_in[26];
    const float* af_p5  = (const float*)d_in[27];
    const float* sr_p1  = (const float*)d_in[28];
    const float* sr_p2  = (const float*)d_in[29];
    const float* sr_p3  = (const float*)d_in[30];
    float* out = (float*)d_out;
    const int E = in_sizes[2];

    // ---- workspace bump allocator ----
    char* ws = (char*)d_ws;
    size_t off = 0;
    auto alloc = [&](size_t bytes) -> void* {
        void* p = ws + off;
        off = (off + bytes + 255) & ~(size_t)255;
        return p;
    };
    auto kpad = [](int ktot) { return ((ktot + 31) / 32) * 32; };

    _Float16* h16A  = (_Float16*)alloc((size_t)BATCH * 64 * HW * 2);
    _Float16* h16B  = (_Float16*)alloc((size_t)BATCH * 64 * HW * 2);
    float*    c1f   = (float*)   alloc((size_t)BATCH * 64 * HW * 4);
    float*    mv    = (float*)   alloc(512);
    float*    nodef = (float*)   alloc((size_t)BATCH * NSN * 8 * 4);
    float*    xl    = (float*)   alloc((size_t)BATCH * NSN * 10 * 4);
    float*    deg   = (float*)   alloc((size_t)2 * NSN * 4);
    float*    dis   = (float*)   alloc((size_t)2 * NSN * 4);
    float*    agg   = (float*)   alloc((size_t)BATCH * 10 * HW * 4);
    _Float16* wp1   = (_Float16*)alloc((size_t)kpad(49)   * 64 * 2);
    _Float16* wp2   = (_Float16*)alloc((size_t)kpad(3136) * 64 * 2);
    _Float16* wp3   = (_Float16*)alloc((size_t)kpad(3136) * 64 * 2);
    _Float16* wp4   = (_Float16*)alloc((size_t)kpad(3136) * 64 * 2);
    _Float16* wp5   = (_Float16*)alloc((size_t)kpad(3136) * 16 * 2);
    _Float16* wps1  = (_Float16*)alloc((size_t)kpad(810)  * 64 * 2);
    _Float16* wps2  = (_Float16*)alloc((size_t)kpad(64)   * 32 * 2);
    _Float16* wps3  = (_Float16*)alloc((size_t)kpad(800)  * 16 * 2);

    auto g1 = [](int n) { return dim3((n + 255) / 256); };
    dim3 cgrid(8, 128, BATCH);

    #define PACK(w_, wp_, Ktot_, Cout_, CoutPad_)                                        \
        do {                                                                             \
            int nt_ = (CoutPad_) / 16;                                                   \
            int tf_ = (((Ktot_) + 31) / 32) * nt_ * 512;                                 \
            pack_w_frag_k<<<g1(tf_), 256, 0, stream>>>(w_, wp_, Ktot_, Cout_, nt_, tf_); \
        } while (0)

    #define CONV(KS, act_, wp_, bias_, pr_, Cin_, Cout_, CoutPad_, mode_, oh_, of_)      \
        conv_wmma_k<KS><<<cgrid, dim3(32 * ((CoutPad_) / 16)),                           \
                          (size_t)(KS) * (16 + (KS) - 1) * (Cin_) * sizeof(_Float16),    \
                          stream>>>(act_, (const v16h*)(wp_), bias_, pr_,                \
                                    Cin_, Cout_, CoutPad_, mode_, oh_, of_)

    // ---- pack all weight matrices into B-fragment order ----
    PACK(af_w1, wp1,  49,   64, 64);
    PACK(af_w2, wp2,  3136, 64, 64);
    PACK(af_w3, wp3,  3136, 64, 64);
    PACK(af_w4, wp4,  3136, 64, 64);
    PACK(af_w5, wp5,  3136,  8, 16);
    PACK(sr_w1, wps1, 810,  64, 64);
    PACK(sr_w2, wps2, 64,   32, 32);
    PACK(sr_w3, wps3, 800,   1, 16);

    // ---- sinogram CNN ----
    cast_f16_k<<<g1(BATCH * HW), 256, 0, stream>>>(sino, h16A, BATCH * HW);
    CONV(7, h16A, wp1, af_b1, af_p1,  1, 64, 64, /*f32*/1, (_Float16*)nullptr, c1f);
    bn_reduce_k<<<64, 256, 0, stream>>>(c1f, mv);
    bn_norm_k<<<g1(BATCH * 64 * HW), 256, 0, stream>>>(c1f, mv, bn_g, bn_b, h16B);
    CONV(7, h16B, wp2, af_b2, af_p2, 64, 64, 64, 0, h16A, (float*)nullptr);
    CONV(7, h16A, wp3, af_b3, af_p3, 64, 64, 64, 0, h16B, (float*)nullptr);
    CONV(7, h16B, wp4, af_b4, af_p4, 64, 64, 64, 0, h16A, (float*)nullptr);
    CONV(7, h16A, wp5, af_b5, af_p5, 64,  8, 16, /*nodefeat*/2, (_Float16*)nullptr, nodef);

    // ---- GCNConv(8 -> 10) ----
    node_lin_k<<<g1(BATCH * NSN), 256, 0, stream>>>(nodef, gcn_w, xl);
    deg_init_k<<<g1(2 * NSN), 256, 0, stream>>>(deg);
    deg_acc_k<<<g1(E), 256, 0, stream>>>(ei, ew, deg, E);
    dis_k<<<g1(2 * NSN), 256, 0, stream>>>(deg, dis);
    zero_k<<<g1(BATCH * 10 * HW), 256, 0, stream>>>(agg, BATCH * 10 * HW);
    scatter_k<<<g1(4 * E), 256, 0, stream>>>(ei, ew, dis, xl, agg, E);
    img_cast_k<<<g1(BATCH * 10 * HW), 256, 0, stream>>>(agg, gcn_b, h16A);

    // ---- SRNET ----
    CONV(9, h16A, wps1, sr_b1, sr_p1, 10, 64, 64, 0, h16B, (float*)nullptr);
    CONV(1, h16B, wps2, sr_b2, sr_p2, 64, 32, 32, 0, h16A, (float*)nullptr);
    CONV(5, h16A, wps3, sr_b3, sr_p3, 32,  1, 16, /*final*/3, (_Float16*)nullptr, out);
    #undef CONV
    #undef PACK
}
